// BayesKerasDense_38302518346152
// MI455X (gfx1250) — matrix operand order repaired and checked by reading, not compiled, not run on hardware
//
#include <hip/hip_runtime.h>

typedef __attribute__((ext_vector_type(16))) _Float16 v16h;
typedef __attribute__((ext_vector_type(8)))  float    v8f;

#define TT_BATCH  4096
#define TT_NIN    4096
#define TT_NOUT   4096
#define T_FLOATS  65536          // intermediate tensor per batch row (fp32)
#define B_HALFS   (128*128)      // staged core fragments (fp16)

// A-fragment K index for half j within a 16x32 fp16 A tile (matches B staging):
// lanes 0-15 (hi=0): halves 0..15 -> K 0..15
// lanes 16-31 (hi=1): halves 0..15 -> K 16..31
__device__ __forceinline__ int a_kofj(int j, int hi) {
  return hi * 16 + j;
}

// Stage core (fp32, [Kvalid x Nvalid] row-major) into LDS as fp16 WMMA
// B-fragments, padded to 128x128. Fragment order:
//   f = ((kb*8 + nt)*32 + lane)*16 + j,  k = kb*32 + (lane>>4)*16 + j,
//   n = nt*16 + (lane&15)
// so a B tile load is one contiguous 32-byte LDS read per lane.
__device__ __forceinline__ void stage_core(const float* __restrict__ core,
                                           _Float16* __restrict__ Bst,
                                           int Kvalid, int Nvalid, int tid) {
  for (int f = tid; f < B_HALFS; f += 512) {
    int j    = f & 15;
    int lane = (f >> 4) & 31;
    int tile = f >> 9;            // kb*8 + nt
    int kb = tile >> 3, nt = tile & 7;
    int k = kb * 32 + (lane >> 4) * 16 + j;
    int n = nt * 16 + (lane & 15);
    float v = 0.0f;
    if (k < Kvalid && n < Nvalid) v = core[k * Nvalid + n];
    Bst[f] = (_Float16)v;
  }
}

__device__ __forceinline__ v16h load_btile(const _Float16* __restrict__ Bst,
                                           int kb, int nt, int lane) {
  return *(const v16h*)(Bst + ((((kb * 8 + nt) * 32) + lane) << 4));
}

// Middle TT step, in place in LDS: rows (p,s), K=N=128, S = 1<<LOG2S.
// kb-outer / nt-inner ordering: 8 independent WMMAs between dependent ones.
template <int LOG2S>
__device__ __forceinline__ void tt_mid_step(float* __restrict__ t,
                                            const _Float16* __restrict__ Bst,
                                            int lane, int wave) {
  const int hi = lane >> 4, m16 = lane & 15;
  const int S = 1 << LOG2S;
  for (int rtp = 0; rtp < 2; ++rtp) {
    const int rt   = wave + rtp * 16;
    const int rowA = rt * 16 + m16;
    const int pA = rowA >> LOG2S, sA = rowA & (S - 1);
    const int baseA = ((pA << 7) << LOG2S) + sA;   // (p*128)*S + s

    v16h a[4];
#pragma unroll
    for (int kb = 0; kb < 4; ++kb) {
#pragma unroll
      for (int j = 0; j < 16; ++j) {
        int K = kb * 32 + a_kofj(j, hi);
        a[kb][j] = (_Float16)t[baseA + (K << LOG2S)];
      }
    }

    v8f c[8];
#pragma unroll
    for (int nt = 0; nt < 8; ++nt) c[nt] = (v8f){0.f,0.f,0.f,0.f,0.f,0.f,0.f,0.f};
#pragma unroll
    for (int kb = 0; kb < 4; ++kb) {
#pragma unroll
      for (int nt = 0; nt < 8; ++nt) {
        v16h b = load_btile(Bst, kb, nt, lane);
        c[nt] = __builtin_amdgcn_wmma_f32_16x16x32_f16(
            false, a[kb], false, b, (short)0, c[nt], false, false);
      }
    }

    // in-place writeback: rows own their address set, so no intra-step race
#pragma unroll
    for (int nt = 0; nt < 8; ++nt) {
#pragma unroll
      for (int v = 0; v < 8; ++v) {
        int mm   = v + 8 * hi;
        int rowO = rt * 16 + mm;
        int pO = rowO >> LOG2S, sO = rowO & (S - 1);
        int n = nt * 16 + m16;
        t[(((pO << 7) + n) << LOG2S) + sO] = c[nt][v];
      }
    }
  }
}

extern "C" __global__ __launch_bounds__(512)
void tt_dense_relu_kernel(const float* __restrict__ x,
                          const float* __restrict__ c0,
                          const float* __restrict__ c1,
                          const float* __restrict__ c2,
                          const float* __restrict__ c3,
                          const float* __restrict__ bias,
                          float* __restrict__ y) {
  extern __shared__ char smem[];
  float*    t   = (float*)smem;                                 // 256 KB
  _Float16* Bst = (_Float16*)(smem + T_FLOATS * sizeof(float)); // 32 KB

  const int tid  = threadIdx.x;
  const int lane = tid & 31;
  const int wave = tid >> 5;
  const int hi = lane >> 4, m16 = lane & 15;
  const int z = blockIdx.x;
  const float* __restrict__ xrow = x + (size_t)z * TT_NIN;

  // ---- step 0: M=512 (s), K=8 (pad->32), N=128, out t[n*512+s] ----
  stage_core(c0, Bst, 8, 128, tid);
  __syncthreads();
  for (int rtp = 0; rtp < 2; ++rtp) {
    int rt = wave + rtp * 16;
    int s  = rt * 16 + m16;
    // Only hi==0 lanes, halves j<8 carry valid K (=j). Load unconditionally
    // (address always valid), select with cndmask -- no exec-mask branches.
    v16h a;
#pragma unroll
    for (int j = 0; j < 8; ++j) {
      float val = xrow[j * 512 + s];
      a[j] = (hi == 0) ? (_Float16)val : (_Float16)0.f;
    }
#pragma unroll
    for (int j = 8; j < 16; ++j) a[j] = (_Float16)0.f;

    v8f c[8];
#pragma unroll
    for (int nt = 0; nt < 8; ++nt) {
      v8f acc = {0.f, 0.f, 0.f, 0.f, 0.f, 0.f, 0.f, 0.f};
      v16h b = load_btile(Bst, 0, nt, lane);
      c[nt] = __builtin_amdgcn_wmma_f32_16x16x32_f16(
          false, a, false, b, (short)0, acc, false, false);
    }
#pragma unroll
    for (int nt = 0; nt < 8; ++nt) {
#pragma unroll
      for (int v = 0; v < 8; ++v) {
        int mm = v + 8 * hi;
        int srow = rt * 16 + mm;
        int n = nt * 16 + m16;
        t[n * 512 + srow] = c[nt][v];
      }
    }
  }
  __syncthreads();

  // ---- step 1: S=64 ----
  stage_core(c1, Bst, 128, 128, tid);
  __syncthreads();
  tt_mid_step<6>(t, Bst, lane, wave);
  __syncthreads();

  // ---- step 2: S=8 ----
  stage_core(c2, Bst, 128, 128, tid);
  __syncthreads();
  tt_mid_step<3>(t, Bst, lane, wave);
  __syncthreads();

  // ---- step 3: M=512 (p), K=128, N=8 (pad->16), fused bias+relu ----
  stage_core(c3, Bst, 128, 8, tid);
  __syncthreads();
  float* __restrict__ yrow = y + (size_t)z * TT_NOUT;
#pragma unroll
  for (int rtp = 0; rtp < 2; ++rtp) {
    int rt   = wave + rtp * 16;
    int rowA = rt * 16 + m16;
    v16h a[4];
#pragma unroll
    for (int kb = 0; kb < 4; ++kb) {
#pragma unroll
      for (int j = 0; j < 16; ++j) {
        int K = kb * 32 + a_kofj(j, hi);
        a[kb][j] = (_Float16)t[rowA * 128 + K];
      }
    }
    v8f acc = {0.f, 0.f, 0.f, 0.f, 0.f, 0.f, 0.f, 0.f};
#pragma unroll
    for (int kb = 0; kb < 4; ++kb) {
      v16h b = load_btile(Bst, kb, 0, lane);
      acc = __builtin_amdgcn_wmma_f32_16x16x32_f16(
          false, a[kb], false, b, (short)0, acc, false, false);
    }
    if (m16 < 8) {
#pragma unroll
      for (int v = 0; v < 8; ++v) {
        int mm   = v + 8 * hi;
        int rowO = rt * 16 + mm;
        int idx  = rowO * 8 + m16;
        float val = acc[v] + bias[idx];
        yrow[idx] = fmaxf(val, 0.0f);
      }
    }
  }
}

extern "C" void kernel_launch(void* const* d_in, const int* in_sizes, int n_in,
                              void* d_out, int out_size, void* d_ws, size_t ws_size,
                              hipStream_t stream) {
  const float* x  = (const float*)d_in[0];
  const float* c0 = (const float*)d_in[1];
  const float* c1 = (const float*)d_in[2];
  const float* c2 = (const float*)d_in[3];
  const float* c3 = (const float*)d_in[4];
  const float* b  = (const float*)d_in[5];
  float* y = (float*)d_out;

  const size_t shmem = T_FLOATS * sizeof(float) + B_HALFS * sizeof(_Float16); // 288 KB
  hipFuncSetAttribute((const void*)tt_dense_relu_kernel,
                      hipFuncAttributeMaxDynamicSharedMemorySize, (int)shmem);
  tt_dense_relu_kernel<<<TT_BATCH, 512, shmem, stream>>>(x, c0, c1, c2, c3, b, y);
}
